// MultiHeadAttention_35296041239162
// MI455X (gfx1250) — matrix operand order repaired and checked by reading, not compiled, not run on hardware
//
#include <hip/hip_runtime.h>

// ---------------------------------------------------------------------------
// MultiHeadAttention forward for MI455X (gfx1250, wave32, WMMA 16x16x32 f16)
// B=2, S=2048, D=768, H=12, DH=64
//
// Pipeline:
//   0) cvt       : one bandwidth pass converting x/q/k/v and all weights to f16
//   1) proj      : Q/K = x@W.T+b -> f16 [B,H,S,DH];  V -> f16 [B,H,DH,S]
//                  (wave tile 64x64 = 16 WMMA per k-step, pure-f16 operands)
//                  Q is pre-scaled by log2(e)/sqrt(DH) so attention softmax
//                  can run in base-2 with no per-element scale multiplies.
//   2) attn      : flash attention; scores computed K-major (A=K, B=Q) so mask
//                  loads / softmax / LDS staging are contiguous per lane
//   3) out_proj  : ctx@Wo.T+bo -> f32 output
// ---------------------------------------------------------------------------

#define B_   2
#define S_   2048
#define D_   768
#define H_   12
#define DH_  64
#define BSD_ (B_ * S_ * D_)   // 3,145,728 elements
#define DD_  (D_ * D_)        //   589,824 elements

// log2(e) / sqrt(DH): folded into Q and the mask so p = exp2(s' - m')
#define QSCALE_ 0.1803368801111204f

typedef __attribute__((ext_vector_type(16))) _Float16 v16h;
typedef __attribute__((ext_vector_type(8)))  _Float16 v8h;
typedef __attribute__((ext_vector_type(4)))  _Float16 v4h;
typedef __attribute__((ext_vector_type(8)))  float    v8f;

static __device__ __forceinline__ v8f wmma_f16(v16h a, v16h b, v8f c) {
  // (neg_a, A, neg_b, B, c_mod, C, reuse_a, reuse_b)
  return __builtin_amdgcn_wmma_f32_16x16x32_f16(false, a, false, b, (short)0, c,
                                                false, false);
}

// --- Fragment loaders (contiguous k order, same for A and B) -----------
// A (16x32, M x K): lane L -> row m=L&15, hh=L>>4;
//   halves [0..15] = K hh*16 + 0..15.
static __device__ __forceinline__ v16h load_a_f16(const _Float16* X, int row0,
                                                  int ld, int k0, int lane) {
  const int m = lane & 15, hh = lane >> 4;
  const _Float16* p = X + (size_t)(row0 + m) * ld + k0 + hh * 16;
  v16h a;
#pragma unroll
  for (int i = 0; i < 16; ++i) a[i] = p[i];
  return a;
}

// B (32x16, K x N) with B[k,n] = M[n0+n][k0+k] for row-major M (i.e. M^T):
// lane L -> col n=L&15, kh=L>>4; halves [0..15] = K kh*16 + 0..15.
static __device__ __forceinline__ v16h load_b_f16(const _Float16* W, int n0,
                                                  int ld, int k0, int lane) {
  const int n = lane & 15, kh = lane >> 4;
  const _Float16* p = W + (size_t)(n0 + n) * ld + k0 + kh * 16;
  v16h b;
#pragma unroll
  for (int i = 0; i < 16; ++i) b[i] = p[i];
  return b;
}

// ---------------------------------------------------------------------------
// Kernel 0: f32 -> f16 conversion (n4 = elements/4)
// ---------------------------------------------------------------------------
__global__ __launch_bounds__(256)
void cvt_kernel(const float* __restrict__ src, _Float16* __restrict__ dst, int n4) {
  const int i = blockIdx.x * 256 + threadIdx.x;
  if (i < n4) {
    const float4 f = ((const float4*)src)[i];
    v4h h = { (_Float16)f.x, (_Float16)f.y, (_Float16)f.z, (_Float16)f.w };
    ((v4h*)dst)[i] = h;
  }
}

// ---------------------------------------------------------------------------
// Kernel 1: y = (x @ W.T + bias) * oscale (all-f16 operands, f32 accumulate).
// Wave tile 64x64 (4x4 WMMA frags); block = 8 waves (4M x 2N) = 256x128 tile.
// transposeV==0: store [B,H,S,DH];  transposeV==1: store [B,H,DH,S].
// ---------------------------------------------------------------------------
__global__ __launch_bounds__(256)
void proj_kernel(const _Float16* __restrict__ X, const _Float16* __restrict__ W,
                 const float* __restrict__ bias, _Float16* __restrict__ out,
                 int transposeV, float oscale) {
  const int tid = threadIdx.x, wave = tid >> 5, lane = tid & 31;
  const int wm = wave & 3, wn = wave >> 2;
  const int row0 = blockIdx.x * 256 + wm * 64;
  const int col0 = blockIdx.y * 128 + wn * 64;

  v8f acc[4][4] = {};
  for (int k0 = 0; k0 < D_; k0 += 32) {
    v16h a[4], bf[4];
#pragma unroll
    for (int mf = 0; mf < 4; ++mf) a[mf] = load_a_f16(X, row0 + mf * 16, D_, k0, lane);
#pragma unroll
    for (int nf = 0; nf < 4; ++nf) bf[nf] = load_b_f16(W, col0 + nf * 16, D_, k0, lane);
#pragma unroll
    for (int mf = 0; mf < 4; ++mf)
#pragma unroll
      for (int nf = 0; nf < 4; ++nf)
        acc[mf][nf] = wmma_f16(a[mf], bf[nf], acc[mf][nf]);
  }

  const int n = lane & 15, hh = lane >> 4;
#pragma unroll
  for (int mf = 0; mf < 4; ++mf) {
#pragma unroll
    for (int nf = 0; nf < 4; ++nf) {
      const int gn = col0 + nf * 16 + n;
      const int head = gn >> 6, dh = gn & 63;
      const float bval = bias[gn];
      if (transposeV) {
        // [B,H,DH,S]: consecutive C-frag rows are consecutive s -> 16B store
        const int gm0 = row0 + mf * 16 + hh * 8;
        const int bb = gm0 >> 11, sbase = gm0 & (S_ - 1);
        v8h pk;
#pragma unroll
        for (int j = 0; j < 8; ++j) pk[j] = (_Float16)((acc[mf][nf][j] + bval) * oscale);
        *(v8h*)&out[((size_t)(bb * H_ + head) * DH_ + dh) * S_ + sbase] = pk;
      } else {
#pragma unroll
        for (int j = 0; j < 8; ++j) {
          const int gm = row0 + mf * 16 + hh * 8 + j;
          const int bb = gm >> 11, s = gm & (S_ - 1);
          out[((size_t)(bb * H_ + head) * S_ + s) * DH_ + dh] =
              (_Float16)((acc[mf][nf][j] + bval) * oscale);
        }
      }
    }
  }
}

// ---------------------------------------------------------------------------
// Kernel 2: flash attention.  grid.x = B*H, grid.y = S/128; 8 waves/block,
// each wave owns 16 query rows, streaming keys in chunks of 64.
//
// Scores are computed transposed (A = K tile, B = Q tile), so each lane owns
// one query column with contiguous key runs: mask loads are 2xb128 per tile,
// softmax reductions are 1 shfl_xor(16), and the P tile is staged to LDS with
// one 16B ds_store per tile.  Q was pre-scaled by log2(e)/8 and the mask is
// folded in with a single FMA, so the softmax runs in base-2 (raw v_exp_f32).
// ---------------------------------------------------------------------------
__global__ __launch_bounds__(256)
void attn_kernel(const _Float16* __restrict__ Qh, const _Float16* __restrict__ Kh,
                 const _Float16* __restrict__ Vt, const float* __restrict__ mask,
                 _Float16* __restrict__ ctx) {
  __shared__ _Float16 plds[8 * 16 * 64];   // 2KB per wave

  const int tid = threadIdx.x, wave = tid >> 5, lane = tid & 31;
  const int n = lane & 15, hh = lane >> 4;
  const int bh = blockIdx.x;
  const int b = bh / H_, h = bh % H_;
  const int s0 = blockIdx.y * 128 + wave * 16;

  const _Float16* Qb = Qh + (size_t)bh * S_ * DH_;
  const _Float16* Kb = Kh + (size_t)bh * S_ * DH_;
  const _Float16* Vb = Vt + (size_t)bh * DH_ * S_;   // [DH][S]
  _Float16* pl = plds + wave * (16 * 64);

  // Q held as B fragments for the whole loop: B[d,q] = Q[q,d]
  const v16h qb0 = load_b_f16(Qb, s0, DH_, 0,  lane);
  const v16h qb1 = load_b_f16(Qb, s0, DH_, 32, lane);

  v8f oacc[4] = {};                 // 16 q-rows x 64 dh accumulator (f32)
  float rmax = -1e30f, rsum = 0.0f; // per-lane scalars for query q = s0+n
  const float* mrow = mask + (size_t)(s0 + n) * S_;

  for (int j0 = 0; j0 < S_; j0 += 64) {
    // prefetch next chunk of K (8KB contiguous) and V rows (global_prefetch_b8)
    if (j0 + 64 < S_) {
      __builtin_prefetch(Kb + (size_t)(j0 + 64 + (lane << 1)) * DH_, 0, 1);
      __builtin_prefetch(Vb + (size_t)lane * S_ + j0 + 64, 0, 1);
      __builtin_prefetch(Vb + (size_t)(lane + 32) * S_ + j0 + 64, 0, 1);
    }

    // ---- scores^T: 4 tiles of 16 keys; element (key = t*16+hh*8+j, q = n) --
    v8f st[4];
#pragma unroll
    for (int t = 0; t < 4; ++t) {
      v8f c = {};
      c = wmma_f16(load_a_f16(Kb, j0 + t * 16, DH_, 0,  lane), qb0, c);
      c = wmma_f16(load_a_f16(Kb, j0 + t * 16, DH_, 32, lane), qb1, c);
      st[t] = c;
    }

    // ---- s' = QK*qs + mask*qs via one FMA; chunk max (contiguous loads) ---
    float cm = -1e30f;
#pragma unroll
    for (int t = 0; t < 4; ++t) {
      const float* mp = mrow + j0 + t * 16 + hh * 8;
#pragma unroll
      for (int j = 0; j < 8; ++j) {
        const float sv = fmaf(mp[j], QSCALE_, st[t][j]);
        st[t][j] = sv;
        cm = fmaxf(cm, sv);
      }
    }
    cm = fmaxf(cm, __shfl_xor(cm, 16, 32));   // combine key halves

    const float nm = fmaxf(rmax, cm);
    const float fs = exp2f(rmax - nm);        // base-2: raw v_exp_f32
    rmax = nm;

    // ---- p = exp2(s' - max): per-lane sum + 16B LDS stores ([q][k]) -------
    float cs = 0.0f;
#pragma unroll
    for (int t = 0; t < 4; ++t) {
      v8h ph;
#pragma unroll
      for (int j = 0; j < 8; ++j) {
        const float p = exp2f(st[t][j] - nm);
        cs += p;
        ph[j] = (_Float16)p;
      }
      *(v8h*)(pl + n * 64 + t * 16 + hh * 8) = ph;
    }
    cs += __shfl_xor(cs, 16, 32);
    rsum = rsum * fs + cs;

    // ---- broadcast per-query rescale onto accumulator rows ----------------
    float fsb[8];
#pragma unroll
    for (int j = 0; j < 8; ++j) fsb[j] = __shfl(fs, hh * 8 + j, 32);
#pragma unroll
    for (int d = 0; d < 4; ++d)
#pragma unroll
      for (int j = 0; j < 8; ++j) oacc[d][j] *= fsb[j];

    // ---- wait this wave's LDS stores, reload P as A fragments -------------
    asm volatile("s_wait_dscnt 0x0" ::: "memory");
    const v16h pa0 = load_a_f16(pl, 0, 64, 0,  lane);
    const v16h pa1 = load_a_f16(pl, 0, 64, 32, lane);

    // ---- P @ V : V^T layout gives contiguous 32B B-frag loads -------------
#pragma unroll
    for (int d = 0; d < 4; ++d) {
      oacc[d] = wmma_f16(pa0, load_b_f16(Vb, d * 16, S_, j0,      lane), oacc[d]);
      oacc[d] = wmma_f16(pa1, load_b_f16(Vb, d * 16, S_, j0 + 32, lane), oacc[d]);
    }
  }

  // ---- normalize and store ctx [B,S,D] as f16 -----------------------------
  const float inv = 1.0f / (rsum + 1e-20f);
  float invb[8];
#pragma unroll
  for (int j = 0; j < 8; ++j) invb[j] = __shfl(inv, hh * 8 + j, 32);
#pragma unroll
  for (int d = 0; d < 4; ++d) {
    const int gn = h * DH_ + d * 16 + n;
#pragma unroll
    for (int j = 0; j < 8; ++j) {
      const int sq = s0 + hh * 8 + j;
      ctx[((size_t)b * S_ + sq) * D_ + gn] = (_Float16)(oacc[d][j] * invb[j]);
    }
  }
}

// ---------------------------------------------------------------------------
// Kernel 3: out = ctx @ Wo.T + bo  (f16 operands, f32 output)
// ---------------------------------------------------------------------------
__global__ __launch_bounds__(256)
void out_proj_kernel(const _Float16* __restrict__ X, const _Float16* __restrict__ W,
                     const float* __restrict__ bias, float* __restrict__ out) {
  const int tid = threadIdx.x, wave = tid >> 5, lane = tid & 31;
  const int wm = wave & 3, wn = wave >> 2;
  const int row0 = blockIdx.x * 256 + wm * 64;
  const int col0 = blockIdx.y * 128 + wn * 64;

  v8f acc[4][4] = {};
  for (int k0 = 0; k0 < D_; k0 += 32) {
    v16h a[4], bf[4];
#pragma unroll
    for (int mf = 0; mf < 4; ++mf) a[mf] = load_a_f16(X, row0 + mf * 16, D_, k0, lane);
#pragma unroll
    for (int nf = 0; nf < 4; ++nf) bf[nf] = load_b_f16(W, col0 + nf * 16, D_, k0, lane);
#pragma unroll
    for (int mf = 0; mf < 4; ++mf)
#pragma unroll
      for (int nf = 0; nf < 4; ++nf)
        acc[mf][nf] = wmma_f16(a[mf], bf[nf], acc[mf][nf]);
  }

  const int n = lane & 15, hh = lane >> 4;
#pragma unroll
  for (int mf = 0; mf < 4; ++mf) {
#pragma unroll
    for (int nf = 0; nf < 4; ++nf) {
      const int gn = col0 + nf * 16 + n;
      const float bval = bias[gn];
#pragma unroll
      for (int j = 0; j < 8; ++j) {
        const int gm = row0 + mf * 16 + hh * 8 + j;
        out[(size_t)gm * D_ + gn] = acc[mf][nf][j] + bval;
      }
    }
  }
}

// ---------------------------------------------------------------------------
extern "C" void kernel_launch(void* const* d_in, const int* in_sizes, int n_in,
                              void* d_out, int out_size, void* d_ws, size_t ws_size,
                              hipStream_t stream) {
  const float* q    = (const float*)d_in[0];
  const float* k    = (const float*)d_in[1];
  const float* v    = (const float*)d_in[2];
  const float* mask = (const float*)d_in[3];
  const float* Wq   = (const float*)d_in[4];
  const float* bq   = (const float*)d_in[5];
  const float* Wk   = (const float*)d_in[6];
  const float* bk   = (const float*)d_in[7];
  const float* Wv   = (const float*)d_in[8];
  const float* bv   = (const float*)d_in[9];
  const float* Wo   = (const float*)d_in[10];
  const float* bo   = (const float*)d_in[11];
  float* out = (float*)d_out;

  // Workspace layout (f16): ~48.8 MB total
  _Float16* ws  = (_Float16*)d_ws;
  _Float16* Xq  = ws;               // BSD
  _Float16* Xk  = Xq  + BSD_;
  _Float16* Xv  = Xk  + BSD_;
  _Float16* Whq = Xv  + BSD_;       // DD
  _Float16* Whk = Whq + DD_;
  _Float16* Whv = Whk + DD_;
  _Float16* Who = Whv + DD_;
  _Float16* Qh  = Who + DD_;        // BSD
  _Float16* Kh  = Qh  + BSD_;
  _Float16* Vt  = Kh  + BSD_;
  _Float16* Ctx = Vt  + BSD_;

  // ---- one-shot f32 -> f16 conversion pass --------------------------------
  const int bsd4 = BSD_ / 4, dd4 = DD_ / 4;
  cvt_kernel<<<(bsd4 + 255) / 256, 256, 0, stream>>>(q,  Xq,  bsd4);
  cvt_kernel<<<(bsd4 + 255) / 256, 256, 0, stream>>>(k,  Xk,  bsd4);
  cvt_kernel<<<(bsd4 + 255) / 256, 256, 0, stream>>>(v,  Xv,  bsd4);
  cvt_kernel<<<(dd4  + 255) / 256, 256, 0, stream>>>(Wq, Whq, dd4);
  cvt_kernel<<<(dd4  + 255) / 256, 256, 0, stream>>>(Wk, Whk, dd4);
  cvt_kernel<<<(dd4  + 255) / 256, 256, 0, stream>>>(Wv, Whv, dd4);
  cvt_kernel<<<(dd4  + 255) / 256, 256, 0, stream>>>(Wo, Who, dd4);

  // ---- projections: M=4096/256, N=768/128 ---------------------------------
  // Q carries log2(e)/sqrt(DH) so attention softmax runs in base-2.
  dim3 gproj(16, 6);
  proj_kernel<<<gproj, 256, 0, stream>>>(Xq, Whq, bq, Qh, 0, QSCALE_);
  proj_kernel<<<gproj, 256, 0, stream>>>(Xk, Whk, bk, Kh, 0, 1.0f);
  proj_kernel<<<gproj, 256, 0, stream>>>(Xv, Whv, bv, Vt, 1, 1.0f);

  // ---- attention ----------------------------------------------------------
  attn_kernel<<<dim3(B_ * H_, S_ / 128), 256, 0, stream>>>(Qh, Kh, Vt, mask, Ctx);

  // ---- output projection --------------------------------------------------
  out_proj_kernel<<<gproj, 256, 0, stream>>>(Ctx, Who, bo, out);
}
